// NCutLoss2D_43963285241856
// MI455X (gfx1250) — matrix operand order repaired and checked by reading, not compiled, not run on hardware
//
#include <hip/hip_runtime.h>
#include <math.h>

typedef __attribute__((ext_vector_type(16))) _Float16 v16h;
typedef __attribute__((ext_vector_type(8)))  float    v8f;

#define B_  16
#define K_  20
#define H_  256
#define W_  256
#define HW  (H_ * W_)

// ---------------------------------------------------------------------------
// Kernel 1: per-(b,k) class-mean stats.
// class_mean_c = sum(l*x_c) / (sum(l) + 1e-5*HW)
// ---------------------------------------------------------------------------
__global__ __launch_bounds__(256) void ncut_stats(
    const float* __restrict__ labels, const float* __restrict__ inputs,
    float* __restrict__ cm4)
{
  const int p = blockIdx.x;          // p = b*K + k
  const int b = p / K_;
  const float* lab = labels + (size_t)p * HW;
  const float* inp = inputs + (size_t)b * 3 * HW;

  float sl = 0.f, s0 = 0.f, s1 = 0.f, s2 = 0.f;
  for (int i = threadIdx.x; i < HW; i += 256) {
    float l = lab[i];
    sl += l;
    s0 += l * inp[i];
    s1 += l * inp[i + HW];
    s2 += l * inp[i + 2 * HW];
  }
  __shared__ float red[4][256];
  red[0][threadIdx.x] = sl; red[1][threadIdx.x] = s0;
  red[2][threadIdx.x] = s1; red[3][threadIdx.x] = s2;
  __syncthreads();
  for (int s = 128; s > 0; s >>= 1) {
    if (threadIdx.x < s) {
      #pragma unroll
      for (int q = 0; q < 4; ++q)
        red[q][threadIdx.x] += red[q][threadIdx.x + s];
    }
    __syncthreads();
  }
  if (threadIdx.x == 0) {
    float denom = red[0][0] + 1e-5f * (float)HW;
    float c0 = red[1][0] / denom, c1 = red[2][0] / denom, c2 = red[3][0] / denom;
    cm4[p * 4 + 0] = c0; cm4[p * 4 + 1] = c1;
    cm4[p * 4 + 2] = c2; cm4[p * 4 + 3] = 0.f;
  }
}

// ---------------------------------------------------------------------------
// Separable 9-tap blur of one 16x16 tile via 3x v_wmma_f32_16x16x32_f16.
//   pass1: tmp^T = X^T(16x32) * G(32x16)   (vertical blur, transposed out)
//   pass2: out   = tmp(16x32) * G(32x16)   (horizontal blur)
// C-layout of pass1 feeds A-layout of pass2 with pure in-lane f32->f16 packs.
// src is column-major f16 LDS, 80 rows per column, zero-padded halo.
// ---------------------------------------------------------------------------
__device__ inline v8f blur_tile(const _Float16* src, v16h Gmat,
                                int R0, int C0, int n, int hi)
{
  v16h Aa, Ab;
  #pragma unroll
  for (int v = 0; v < 8; ++v) {
    // A-layout K base: lanes>=16 offset 8, VGPRs 4..7 offset 16
    int Kb = ((v & 4) ? 16 : 0) + hi * 8 + 2 * (v & 3);
    int ra = R0 + Kb;
    const _Float16* pa = src + (C0 + n) * 80 + ra;        // tile a: cols C0..C0+15
    const _Float16* pb = src + (C0 + 16 + n) * 80 + ra;   // tile b: cols C0+16..31
    Aa[2 * v]     = pa[0];
    Aa[2 * v + 1] = pa[1];
    Ab[2 * v]     = pb[0];
    Ab[2 * v + 1] = pb[1];
  }
  v8f zero = {};
  v8f Ca = __builtin_amdgcn_wmma_f32_16x16x32_f16(false, Aa, false, Gmat,
                                                  (short)0, zero, false, false);
  v8f Cb = __builtin_amdgcn_wmma_f32_16x16x32_f16(false, Ab, false, Gmat,
                                                  (short)0, zero, false, false);
  // In-lane repack: A2 K-pattern == concat(Ca, Cb) elementwise per lane.
  v16h A2;
  #pragma unroll
  for (int e = 0; e < 8; ++e) {
    A2[e]     = (_Float16)Ca[e];
    A2[8 + e] = (_Float16)Cb[e];
  }
  return __builtin_amdgcn_wmma_f32_16x16x32_f16(false, A2, false, Gmat,
                                                (short)0, zero, false, false);
}

// ---------------------------------------------------------------------------
// Kernel 2 (fused): weights -> blur(WMMA) -> labels-weighted reduction.
// Block = 64x64 region of one (b,k) plane; 256 threads = 8 waves.
// ---------------------------------------------------------------------------
__global__ __launch_bounds__(256) void ncut_blur_reduce(
    const float* __restrict__ labels, const float* __restrict__ inputs,
    const float* __restrict__ cm4, float* __restrict__ part)
{
  const int p   = blockIdx.z;       // b*K + k
  const int b   = p / K_;
  const int ry0 = blockIdx.y * 64;
  const int cx0 = blockIdx.x * 64;

  __shared__ _Float16 wT [80 * 80];   // weights, column-major [col][row]
  __shared__ _Float16 lwT[80 * 80];   // labels*weights
  __shared__ float    labT[80 * 80];  // labels (f32, for final reduction)
  __shared__ float    redN[256];
  __shared__ float    redD[256];

  const float c0 = cm4[p * 4 + 0];
  const float c1 = cm4[p * 4 + 1];
  const float c2 = cm4[p * 4 + 2];
  const float* lab = labels + (size_t)p * HW;
  const float* inp = inputs + (size_t)b * 3 * HW;

  // ---- Phase 1: build zero-padded 72x72 halo inside an 80x80 LDS pad ----
  for (int idx = threadIdx.x; idx < 80 * 80; idx += 256) {
    int hc = idx / 80, hr = idx % 80;            // stored at [hc*80 + hr]
    float w = 0.f, l = 0.f;
    if (hr < 72 && hc < 72) {
      int r = ry0 - 4 + hr;
      int c = cx0 - 4 + hc;
      if (r >= 0 && r < H_ && c >= 0 && c < W_) {
        int off = r * W_ + c;
        l = lab[off];
        float x0 = inp[off], x1 = inp[off + HW], x2 = inp[off + 2 * HW];
        float d = (x0 - c0) * (x0 - c0) + (x1 - c1) * (x1 - c1)
                + (x2 - c2) * (x2 - c2);
        w = __expf(-d * d);                      // exp(-diff^2), sigma2 = 1
      }
    }
    wT [idx] = (_Float16)w;
    lwT[idx] = (_Float16)(l * w);
    labT[idx] = l;
  }
  __syncthreads();

  // ---- Constant banded Gaussian B-matrix: G[t][n] = g[t-n], g[d]=exp(-(d-4)^2/50)
  const int lane = threadIdx.x & 31;
  const int n    = lane & 15;
  const int hi   = lane >> 4;
  v16h Gmat;
  #pragma unroll
  for (int e = 0; e < 16; ++e) {
    int Kk = (e < 8) ? (hi * 8 + e) : (16 + hi * 8 + (e - 8)); // B-layout K index
    int d  = Kk - n;
    float dd = (float)(d - 4);
    float gv = __expf(-dd * dd * 0.02f);
    Gmat[e] = (_Float16)((d >= 0 && d <= 8) ? gv : 0.f);
  }

  // ---- Phase 2: 8 waves x 2 tiles; 2 fields x 3 WMMAs per tile ----
  const int wave = threadIdx.x >> 5;
  float numAcc = 0.f, denAcc = 0.f;
  #pragma unroll
  for (int rep = 0; rep < 2; ++rep) {
    int tile = wave * 2 + rep;                   // 0..15
    int R0 = (tile >> 2) * 16;
    int C0 = (tile & 3) * 16;

    v8f outW  = blur_tile(wT,  Gmat, R0, C0, n, hi);   // blur(weights)
    v8f outLW = blur_tile(lwT, Gmat, R0, C0, n, hi);   // blur(labels*weights)

    // lane holds out(row R0 + v + 8*hi, col C0 + n); LDS coords are +4.
    const float* lp = labT + (C0 + n + 4) * 80 + (R0 + 8 * hi + 4);
    #pragma unroll
    for (int v = 0; v < 8; ++v) {
      float l = lp[v];
      denAcc += l * outW[v];
      numAcc += l * outLW[v];
    }
  }

  // ---- Deterministic block reduction -> per-block partial (no atomics) ----
  redN[threadIdx.x] = numAcc;
  redD[threadIdx.x] = denAcc;
  __syncthreads();
  for (int s = 128; s > 0; s >>= 1) {
    if (threadIdx.x < s) {
      redN[threadIdx.x] += redN[threadIdx.x + s];
      redD[threadIdx.x] += redD[threadIdx.x + s];
    }
    __syncthreads();
  }
  if (threadIdx.x == 0) {
    int pb = blockIdx.z * 16 + blockIdx.y * 4 + blockIdx.x;
    part[pb * 2 + 0] = redN[0];
    part[pb * 2 + 1] = redD[0];
  }
}

// ---------------------------------------------------------------------------
// Kernel 3: deterministic final reduction -> scalar loss.
// ---------------------------------------------------------------------------
__global__ void ncut_final(const float* __restrict__ part, float* __restrict__ out)
{
  __shared__ float terms[32];
  int k = threadIdx.x;
  float t = 0.f;
  if (k < K_) {
    float num = 0.f, den = 0.f;
    for (int bb = 0; bb < B_; ++bb) {
      for (int reg = 0; reg < 16; ++reg) {
        int pb = (bb * K_ + k) * 16 + reg;
        num += part[pb * 2 + 0];
        den += part[pb * 2 + 1];
      }
    }
    t = fabsf(num / (den + 1e-6f));
  }
  terms[threadIdx.x] = t;
  __syncthreads();
  if (threadIdx.x == 0) {
    float loss = 0.f;
    #pragma unroll
    for (int i = 0; i < 32; ++i) loss += terms[i];
    out[0] = (float)K_ - loss;
  }
}

// ---------------------------------------------------------------------------
extern "C" void kernel_launch(void* const* d_in, const int* in_sizes, int n_in,
                              void* d_out, int out_size, void* d_ws, size_t ws_size,
                              hipStream_t stream)
{
  (void)in_sizes; (void)n_in; (void)out_size; (void)ws_size;
  const float* labels = (const float*)d_in[0];  // [16,20,256,256]
  const float* inputs = (const float*)d_in[1];  // [16,3,256,256]
  float* cm4  = (float*)d_ws;                   // 320*4 floats
  float* part = cm4 + (size_t)B_ * K_ * 4;      // 5120*2 floats
  float* out  = (float*)d_out;

  ncut_stats<<<B_ * K_, 256, 0, stream>>>(labels, inputs, cm4);

  dim3 grid(W_ / 64, H_ / 64, B_ * K_);         // (4, 4, 320)
  ncut_blur_reduce<<<grid, 256, 0, stream>>>(labels, inputs, cm4, part);

  ncut_final<<<1, 32, 0, stream>>>(part, out);
}
